// S5_35553739276311
// MI455X (gfx1250) — matrix operand order repaired and checked by reading, not compiled, not run on hardware
//
#include <hip/hip_runtime.h>
#include <hip/hip_bf16.h>
#include <math.h>

// ---------------------------------------------------------------------------
// S5 SSM layer for MI455X (gfx1250, wave32, WMMA).
//   L=32768, H=512, P=256 (complex), N = 2P = 512 = H = K for both GEMMs.
// Pipeline:
//   1) k_params  : bilinear discretization -> Lam_bar, pack W1 (B_bar) and
//                  W2 (2*C_re | -2*C_im) as bf16 hi/lo split weights.
//   2) k_pows    : power table Lam_bar^(j+1), j=0..127 (chunk length 128).
//   3) k_cvt     : u fp32 -> bf16 hi/lo planes.
//   4) k_gemm<0> : Bu = u @ W1   (bf16x3 split WMMA, fp32 accum) -> X ws.
//   5) k_scan_local : per-chunk linear recurrence (256 chunks x 256 ch).
//   6) k_scan_carry : sequential 256-step carry scan (1 block).
//   7) k_scan_fix   : x[l] += Lam^(j+1) * carry_in[chunk]  (streaming).
//   8) k_cvt     : X fp32 -> bf16 hi/lo (reuses u planes).
//   9) k_gemm<1> : y = X @ W2 + d*u -> d_out.
// ---------------------------------------------------------------------------

#define L_SEQ 32768
#define H_FEAT 512
#define P_STATE 256
#define NDIM 512        // 2*P_STATE == H_FEAT
#define KDIM 512
#define CHUNK 128
#define NCHUNK (L_SEQ / CHUNK)

typedef __attribute__((ext_vector_type(16))) __bf16 v16bf;
typedef __attribute__((ext_vector_type(8)))  __bf16 v8bf;
typedef __attribute__((ext_vector_type(8)))  float  v8f;

// ---------------- bf16 helpers (round-to-nearest-even, bit level) ----------
__device__ __forceinline__ __bf16 f2bf(float f) {
    unsigned u = __builtin_bit_cast(unsigned, f);
    unsigned r = u + 0x7FFFu + ((u >> 16) & 1u);
    unsigned short s = (unsigned short)(r >> 16);
    return __builtin_bit_cast(__bf16, s);
}
__device__ __forceinline__ float bf2f(__bf16 b) {
    unsigned short s = __builtin_bit_cast(unsigned short, b);
    unsigned u = ((unsigned)s) << 16;
    return __builtin_bit_cast(float, u);
}
__device__ __forceinline__ void split_bf(float f, __bf16* hi, __bf16* lo) {
    __bf16 h = f2bf(f);
    *hi = h;
    *lo = f2bf(f - bf2f(h));
}

// ---------------- 1) discretization + weight packing -----------------------
__global__ void k_params(const float* __restrict__ lam_re,
                         const float* __restrict__ lam_im,
                         const float* __restrict__ b,      // [P][H][2]
                         const float* __restrict__ c,      // [H][P][2]
                         const float* __restrict__ delta,  // [P]
                         float* __restrict__ a_re, float* __restrict__ a_im,
                         __bf16* __restrict__ W1h, __bf16* __restrict__ W1l, // [K=H][N=2P]
                         __bf16* __restrict__ W2h, __bf16* __restrict__ W2l) // [K=2P][H]
{
    int p = threadIdx.x;                       // 0..255
    float lr = lam_re[p], li = lam_im[p];
    float step = expf(delta[p]);
    float s2 = 0.5f * step;
    // BL = 1 / (1 - s2*Lam)
    float dr = 1.0f - s2 * lr, di = -s2 * li;
    float inv = 1.0f / (dr * dr + di * di);
    float blr = dr * inv, bli = -di * inv;
    // Lam_bar = BL * (1 + s2*Lam)
    float nr = 1.0f + s2 * lr, ni = s2 * li;
    a_re[p] = blr * nr - bli * ni;
    a_im[p] = blr * ni + bli * nr;
    // B_bar scale = BL * step
    float sr = blr * step, si = bli * step;
    for (int h = 0; h < H_FEAT; ++h) {
        float br = b[(p * H_FEAT + h) * 2 + 0];
        float bi = b[(p * H_FEAT + h) * 2 + 1];
        float Br = sr * br - si * bi;
        float Bi = sr * bi + si * br;
        split_bf(Br, &W1h[h * NDIM + p],           &W1l[h * NDIM + p]);
        split_bf(Bi, &W1h[h * NDIM + P_STATE + p], &W1l[h * NDIM + P_STATE + p]);
        float cr = c[(h * P_STATE + p) * 2 + 0];
        float ci = c[(h * P_STATE + p) * 2 + 1];
        split_bf( 2.0f * cr, &W2h[p * H_FEAT + h],             &W2l[p * H_FEAT + h]);
        split_bf(-2.0f * ci, &W2h[(P_STATE + p) * H_FEAT + h], &W2l[(P_STATE + p) * H_FEAT + h]);
    }
}

// ---------------- 2) power table Lam_bar^(j+1) ------------------------------
__global__ void k_pows(const float* __restrict__ a_re, const float* __restrict__ a_im,
                       float* __restrict__ pw_re, float* __restrict__ pw_im) // [CHUNK][P]
{
    int p = threadIdx.x;
    float ar = a_re[p], ai = a_im[p];
    float pr = ar, pi = ai;
    for (int j = 0; j < CHUNK; ++j) {
        pw_re[j * P_STATE + p] = pr;
        pw_im[j * P_STATE + p] = pi;
        float nr = pr * ar - pi * ai;
        float ni = pr * ai + pi * ar;
        pr = nr; pi = ni;
    }
}

// ---------------- 3/8) fp32 -> bf16 hi/lo planes ----------------------------
__global__ void k_cvt(const float* __restrict__ src, __bf16* __restrict__ hi,
                      __bf16* __restrict__ lo, int n4)
{
    int i = blockIdx.x * blockDim.x + threadIdx.x;
    if (i >= n4) return;
    float4 v = ((const float4*)src)[i];
    int base = i * 4;
    split_bf(v.x, &hi[base + 0], &lo[base + 0]);
    split_bf(v.y, &hi[base + 1], &lo[base + 1]);
    split_bf(v.z, &hi[base + 2], &lo[base + 2]);
    split_bf(v.w, &hi[base + 3], &lo[base + 3]);
}

// ---------------- WMMA fragment assembly from LDS rows ----------------------
// A 16x32 bf16 layout (ISA 7.12.2): lanes 0-15 row M=lane, K={0..7,16..23};
// lanes 16-31 row M=lane-16, K={8..15,24..31}.  Row of 32 bf16 = 4 chunks of 8.
__device__ __forceinline__ v16bf fragA(const __bf16* row, int half) {
    const v8bf* ch = (const v8bf*)row;
    v8bf x = ch[half];       // K = half*8 .. +7
    v8bf y = ch[half + 2];   // K = half*8+16 .. +7
    return __builtin_shufflevector(x, y, 0,1,2,3,4,5,6,7,8,9,10,11,12,13,14,15);
}
// B 32x16 bf16: lanes 0-15 col N=lane, K={0..7,16..23}; lanes 16-31 col N=lane-16, K={8..15,24..31}.
// LDS B tile stored transposed [n][k] so each fragment reads contiguous K.
__device__ __forceinline__ v16bf fragB(const __bf16* row, int half) {
    const v8bf* ch = (const v8bf*)row;
    v8bf x = ch[half];
    v8bf y = ch[half + 2];
    return __builtin_shufflevector(x, y, 0,1,2,3,4,5,6,7,8,9,10,11,12,13,14,15);
}

// ---------------- 4/9) bf16x3 split WMMA GEMM ------------------------------
// C[L,512] = A[L,512] x B[512,512];  EP=1 adds d[col]*u[row,col] epilogue.
#define BM 128
#define BN 128
#define BK 32

template <int EP>
__global__ __launch_bounds__(256)
void k_gemm(const __bf16* __restrict__ Ahi, const __bf16* __restrict__ Alo,
            const __bf16* __restrict__ Bhi, const __bf16* __restrict__ Blo,
            float* __restrict__ Cout,
            const float* __restrict__ u, const float* __restrict__ dvec)
{
    __shared__ __bf16 Ash[BM][BK];
    __shared__ __bf16 Asl[BM][BK];
    __shared__ __bf16 Bsh[BN][BK];   // transposed [n][k]
    __shared__ __bf16 Bsl[BN][BK];

    const int tid  = threadIdx.x;
    const int lane = tid & 31;
    const int wid  = tid >> 5;           // 8 waves
    const int wm   = wid & 3;            // 4 M-slots of 32 rows
    const int wn   = wid >> 2;           // 2 N-slots of 64 cols
    const int half = lane >> 4;
    const int lrow = lane & 15;

    const int m0 = blockIdx.x * BM;
    const int n0 = blockIdx.y * BN;

    v8f acc[2][4];
#pragma unroll
    for (int t = 0; t < 2; ++t)
#pragma unroll
        for (int s = 0; s < 4; ++s)
#pragma unroll
            for (int r = 0; r < 8; ++r) acc[t][s][r] = 0.0f;

#pragma unroll 1
    for (int kk = 0; kk < KDIM / BK; ++kk) {
        const int k0 = kk * BK;
        // ---- stage A tile (row-major), 8 bf16 (16B) per transaction ----
#pragma unroll
        for (int i = tid; i < (BM * BK) / 8; i += 256) {
            int row = i >> 2;
            int cc  = (i & 3) * 8;
            size_t g = (size_t)(m0 + row) * KDIM + k0 + cc;
            *(uint4*)&Ash[row][cc] = *(const uint4*)(Ahi + g);
            *(uint4*)&Asl[row][cc] = *(const uint4*)(Alo + g);
        }
        // ---- stage B tile transposed to [n][k] ----
#pragma unroll
        for (int i = tid; i < (BK * BN) / 8; i += 256) {
            int k  = i >> 4;
            int nc = (i & 15) * 8;
            size_t g = (size_t)(k0 + k) * NDIM + n0 + nc;
            uint4 vh = *(const uint4*)(Bhi + g);
            uint4 vl = *(const uint4*)(Blo + g);
            __bf16 th[8]; *(uint4*)th = vh;
            __bf16 tl[8]; *(uint4*)tl = vl;
#pragma unroll
            for (int e = 0; e < 8; ++e) {
                Bsh[nc + e][k] = th[e];
                Bsl[nc + e][k] = tl[e];
            }
        }
        // prefetch next A tile into cache (global_prefetch_b8)
        if (kk + 1 < KDIM / BK) {
            size_t g = (size_t)(m0 + (tid >> 1)) * KDIM + (kk + 1) * BK + (tid & 1) * 16;
            __builtin_prefetch(Ahi + g, 0, 1);
        }
        __syncthreads();

        v16bf ah[2], al[2];
#pragma unroll
        for (int t = 0; t < 2; ++t) {
            int r = wm * 32 + t * 16 + lrow;
            ah[t] = fragA(&Ash[r][0], half);
            al[t] = fragA(&Asl[r][0], half);
        }
        v16bf bh[4], bl[4];
#pragma unroll
        for (int s = 0; s < 4; ++s) {
            int n = wn * 64 + s * 16 + lrow;
            bh[s] = fragB(&Bsh[n][0], half);
            bl[s] = fragB(&Bsl[n][0], half);
        }
#pragma unroll
        for (int t = 0; t < 2; ++t)
#pragma unroll
            for (int s = 0; s < 4; ++s) {
                acc[t][s] = __builtin_amdgcn_wmma_f32_16x16x32_bf16(
                    false, ah[t], false, bh[s], (short)0, acc[t][s], false, false);
                acc[t][s] = __builtin_amdgcn_wmma_f32_16x16x32_bf16(
                    false, ah[t], false, bl[s], (short)0, acc[t][s], false, false);
                acc[t][s] = __builtin_amdgcn_wmma_f32_16x16x32_bf16(
                    false, al[t], false, bh[s], (short)0, acc[t][s], false, false);
            }
        __syncthreads();
    }

    // ---- store C; v8f element r -> row = r + 8*half, col = lane%16 ----
#pragma unroll
    for (int t = 0; t < 2; ++t)
#pragma unroll
        for (int s = 0; s < 4; ++s) {
            int gcol = n0 + wn * 64 + s * 16 + lrow;
#pragma unroll
            for (int r = 0; r < 8; ++r) {
                int grow = m0 + wm * 32 + t * 16 + r + 8 * half;
                float v = acc[t][s][r];
                if (EP) v += dvec[gcol] * u[(size_t)grow * H_FEAT + gcol];
                Cout[(size_t)grow * NDIM + gcol] = v;
            }
        }
}

// ---------------- 5) per-chunk local scan (in place over X) -----------------
__global__ void k_scan_local(float* __restrict__ X,          // [L][512] re|im
                             const float* __restrict__ a_re,
                             const float* __restrict__ a_im,
                             float* __restrict__ cend)       // [NCHUNK][512]
{
    int c = blockIdx.x;            // chunk
    int p = threadIdx.x;           // channel
    float ar = a_re[p], ai = a_im[p];
    float xr = 0.0f, xi = 0.0f;
#pragma unroll 4
    for (int j = 0; j < CHUNK; ++j) {
        size_t l = (size_t)c * CHUNK + j;
        float br = X[l * NDIM + p];
        float bi = X[l * NDIM + P_STATE + p];
        float nr = fmaf(ar, xr, fmaf(-ai, xi, br));
        float ni = fmaf(ar, xi, fmaf( ai, xr, bi));
        xr = nr; xi = ni;
        X[l * NDIM + p]           = xr;
        X[l * NDIM + P_STATE + p] = xi;
    }
    cend[(size_t)c * NDIM + p]           = xr;
    cend[(size_t)c * NDIM + P_STATE + p] = xi;
}

// ---------------- 6) sequential carry scan over chunks ----------------------
__global__ void k_scan_carry(const float* __restrict__ cend,
                             const float* __restrict__ pw_re,
                             const float* __restrict__ pw_im,
                             float* __restrict__ cin)        // [NCHUNK][512]
{
    int p = threadIdx.x;
    float aLr = pw_re[(CHUNK - 1) * P_STATE + p];   // Lam_bar^CHUNK
    float aLi = pw_im[(CHUNK - 1) * P_STATE + p];
    float sr = 0.0f, si = 0.0f;
    for (int c = 0; c < NCHUNK; ++c) {
        cin[(size_t)c * NDIM + p]           = sr;
        cin[(size_t)c * NDIM + P_STATE + p] = si;
        float er = cend[(size_t)c * NDIM + p];
        float ei = cend[(size_t)c * NDIM + P_STATE + p];
        float nr = fmaf(aLr, sr, fmaf(-aLi, si, er));
        float ni = fmaf(aLr, si, fmaf( aLi, sr, ei));
        sr = nr; si = ni;
    }
}

// ---------------- 7) fix-up: x[l] += Lam^(j+1) * cin[chunk] -----------------
__global__ void k_scan_fix(float* __restrict__ X,
                           const float* __restrict__ pw_re,
                           const float* __restrict__ pw_im,
                           const float* __restrict__ cin)
{
    size_t l = blockIdx.x;         // 0..L-1
    int p = threadIdx.x;
    int c = (int)(l / CHUNK);
    int j = (int)(l % CHUNK);
    if (c == 0) return;            // carry-in is zero for chunk 0
    float pr = pw_re[j * P_STATE + p];
    float pi = pw_im[j * P_STATE + p];
    float cr = cin[(size_t)c * NDIM + p];
    float ci = cin[(size_t)c * NDIM + P_STATE + p];
    X[l * NDIM + p]           += pr * cr - pi * ci;
    X[l * NDIM + P_STATE + p] += pr * ci + pi * cr;
}

// ---------------------------------------------------------------------------
extern "C" void kernel_launch(void* const* d_in, const int* in_sizes, int n_in,
                              void* d_out, int out_size, void* d_ws, size_t ws_size,
                              hipStream_t stream)
{
    const float* u      = (const float*)d_in[0];   // [L][H]
    const float* lam_re = (const float*)d_in[1];   // [P]
    const float* lam_im = (const float*)d_in[2];   // [P]
    const float* b      = (const float*)d_in[3];   // [P][H][2]
    const float* c      = (const float*)d_in[4];   // [H][P][2]
    const float* dvec   = (const float*)d_in[5];   // [H]
    const float* delta  = (const float*)d_in[6];   // [P]
    float* out = (float*)d_out;

    char* ws = (char*)d_ws;
    size_t off = 0;
    auto alloc = [&](size_t bytes) { size_t o = off; off = (off + bytes + 255) & ~(size_t)255; return o; };

    float*  X     = (float*) (ws + alloc((size_t)L_SEQ * NDIM * sizeof(float)));
    __bf16* Ahi   = (__bf16*)(ws + alloc((size_t)L_SEQ * KDIM * sizeof(__bf16)));
    __bf16* Alo   = (__bf16*)(ws + alloc((size_t)L_SEQ * KDIM * sizeof(__bf16)));
    __bf16* W1h   = (__bf16*)(ws + alloc((size_t)KDIM * NDIM * sizeof(__bf16)));
    __bf16* W1l   = (__bf16*)(ws + alloc((size_t)KDIM * NDIM * sizeof(__bf16)));
    __bf16* W2h   = (__bf16*)(ws + alloc((size_t)KDIM * NDIM * sizeof(__bf16)));
    __bf16* W2l   = (__bf16*)(ws + alloc((size_t)KDIM * NDIM * sizeof(__bf16)));
    float*  a_re  = (float*) (ws + alloc(P_STATE * sizeof(float)));
    float*  a_im  = (float*) (ws + alloc(P_STATE * sizeof(float)));
    float*  pw_re = (float*) (ws + alloc((size_t)CHUNK * P_STATE * sizeof(float)));
    float*  pw_im = (float*) (ws + alloc((size_t)CHUNK * P_STATE * sizeof(float)));
    float*  cend  = (float*) (ws + alloc((size_t)NCHUNK * NDIM * sizeof(float)));
    float*  cin   = (float*) (ws + alloc((size_t)NCHUNK * NDIM * sizeof(float)));

    // 1) params + weight packing, 2) power table
    k_params<<<1, P_STATE, 0, stream>>>(lam_re, lam_im, b, c, delta,
                                        a_re, a_im, W1h, W1l, W2h, W2l);
    k_pows<<<1, P_STATE, 0, stream>>>(a_re, a_im, pw_re, pw_im);

    // 3) u -> bf16 hi/lo
    {
        int n4 = (L_SEQ * H_FEAT) / 4;
        k_cvt<<<(n4 + 255) / 256, 256, 0, stream>>>(u, Ahi, Alo, n4);
    }
    // 4) Bu = u @ W1
    {
        dim3 grid(L_SEQ / BM, NDIM / BN);
        k_gemm<0><<<grid, 256, 0, stream>>>(Ahi, Alo, W1h, W1l, X, nullptr, nullptr);
    }
    // 5-7) chunked scan
    k_scan_local<<<NCHUNK, P_STATE, 0, stream>>>(X, a_re, a_im, cend);
    k_scan_carry<<<1, P_STATE, 0, stream>>>(cend, pw_re, pw_im, cin);
    k_scan_fix<<<L_SEQ, P_STATE, 0, stream>>>(X, pw_re, pw_im, cin);

    // 8) X -> bf16 hi/lo (reuse planes)
    {
        int n4 = (L_SEQ * NDIM) / 4;
        k_cvt<<<(n4 + 255) / 256, 256, 0, stream>>>(X, Ahi, Alo, n4);
    }
    // 9) y = X @ W2 + d*u
    {
        dim3 grid(L_SEQ / BM, NDIM / BN);
        k_gemm<1><<<grid, 256, 0, stream>>>(Ahi, Alo, W2h, W2l, out, u, dvec);
    }
}
